// conv2d_80685255623434
// MI455X (gfx1250) — matrix operand order repaired and checked by reading, not compiled, not run on hardware
//
#include <hip/hip_runtime.h>

typedef _Float16 v16h __attribute__((ext_vector_type(16)));
typedef _Float16 v8h  __attribute__((ext_vector_type(8)));
typedef float    v8f  __attribute__((ext_vector_type(8)));

#define IMG_H 4096
#define IMG_W 4096
#define KH 15
#define KW 15
#define OUT_H (IMG_H - KH + 1)   // 4082
#define OUT_W (IMG_W - KW + 1)   // 4082

#define TILE 16
#define WAVES_PER_BLK 8
#define BLK_COLS (WAVES_PER_BLK * TILE)   // 128 output cols per block
#define BLK_ROWS 32                       // 2 stacked 16x16 tiles per wave
#define IN_ROWS (BLK_ROWS + KH - 1)       // 46 input rows per block
#define XPITCH 152                        // f16 units; >=142, mult of 8, 76-dword stride (bank-clean)
#define BNP 40                            // f16 per (kh,n) band row; mult of 8, 20-dword stride
#define BKH (16 * BNP)                    // 640 f16 per kh band tile

__global__ __launch_bounds__(256)
void conv15x15_wmma_f16split(const float* __restrict__ X,
                             const float* __restrict__ Wk,
                             const float* __restrict__ bias,
                             float* __restrict__ out) {
    __shared__ _Float16 sXh[IN_ROWS * XPITCH];   // 13984 B
    __shared__ _Float16 sXl[IN_ROWS * XPITCH];   // 13984 B
    __shared__ _Float16 sBh[KH * BKH];           // 19200 B
    __shared__ _Float16 sBl[KH * BKH];           // 19200 B

    const int tid  = threadIdx.x;
    const int y0   = blockIdx.y * BLK_ROWS;   // first output row of block
    const int xblk = blockIdx.x * BLK_COLS;   // first output col of block

    // ---- Build fragment-ready Toeplitz band tiles, split into f16 hi/lo ----
    // sB*[kh][n][c] = split(W[kh, c-n]), zero outside the band
    for (int i = tid; i < KH * 16 * 32; i += 256) {
        const int kh = i >> 9;             // 512 = 16*32 per kh
        const int n  = (i >> 5) & 15;
        const int c  = i & 31;
        const int idx = c - n;
        float w = Wk[kh * KW + min(max(idx, 0), KW - 1)];
        w = ((unsigned)idx < (unsigned)KW) ? w : 0.0f;
        const _Float16 wh = (_Float16)w;
        const int off = kh * BKH + n * BNP + c;
        sBh[off] = wh;
        sBl[off] = (_Float16)(w - (float)wh);
    }

    // ---- Stage input slab, split into f16 hi/lo (rows y0..y0+45, clamped) ----
    for (int i = tid; i < IN_ROWS * XPITCH; i += 256) {
        const int r  = i / XPITCH;
        const int c  = i - r * XPITCH;
        const int gr = min(y0 + r, IMG_H - 1);
        const int gc = min(xblk + c, IMG_W - 1);
        const float x = X[gr * IMG_W + gc];
        const _Float16 xh = (_Float16)x;
        sXh[i] = xh;
        sXl[i] = (_Float16)(x - (float)xh);
    }
    __syncthreads();

    const int lane = tid & 31;
    const int wave = tid >> 5;             // wave32
    const int l15  = lane & 15;            // A: row m | B,C,D: col n
    const int h    = lane >> 4;            // lane half -> K split
    const int wcol = wave * TILE;          // wave's output-column offset in block

    // Per-lane fragment bases (all kh steps become immediate offsets)
    const _Float16* aBaseH = &sXh[l15 * XPITCH + wcol + 8 * h];
    const _Float16* aBaseL = &sXl[l15 * XPITCH + wcol + 8 * h];
    const _Float16* bBaseH = &sBh[l15 * BNP + 8 * h];
    const _Float16* bBaseL = &sBl[l15 * BNP + 8 * h];

    v8f acc0 = {};   // rows y0 + [0,16)
    v8f acc1 = {};   // rows y0 + [16,32)

    #pragma unroll
    for (int kh = 0; kh < KH; ++kh) {
        // B fragments (32x16 band): lane holds K in [8h,8h+8) and [16+8h,16+8h+8)
        v8h b0 = *(const v8h*)(bBaseH + kh * BKH);
        v8h b1 = *(const v8h*)(bBaseH + kh * BKH + 16);
        v16h bh = __builtin_shufflevector(b0, b1, 0,1,2,3,4,5,6,7,8,9,10,11,12,13,14,15);
        v8h c0 = *(const v8h*)(bBaseL + kh * BKH);
        v8h c1 = *(const v8h*)(bBaseL + kh * BKH + 16);
        v16h bl = __builtin_shufflevector(c0, c1, 0,1,2,3,4,5,6,7,8,9,10,11,12,13,14,15);

        // A fragments (16x32): lane holds K in [8h,8h+8) and [16+8h,16+8h+8)
        #pragma unroll
        for (int t = 0; t < 2; ++t) {      // two stacked row tiles reuse B
            const int roff = (kh + 16 * t) * XPITCH;
            v8h p0 = *(const v8h*)(aBaseH + roff);
            v8h p1 = *(const v8h*)(aBaseH + roff + 16);
            v16h ah = __builtin_shufflevector(p0, p1, 0,1,2,3,4,5,6,7,8,9,10,11,12,13,14,15);
            v8h q0 = *(const v8h*)(aBaseL + roff);
            v8h q1 = *(const v8h*)(aBaseL + roff + 16);
            v16h al = __builtin_shufflevector(q0, q1, 0,1,2,3,4,5,6,7,8,9,10,11,12,13,14,15);

            v8f acc = t ? acc1 : acc0;
            // hi*hi + hi*lo + lo*hi  (lo*lo ~ 2^-22, dropped)
            acc = __builtin_amdgcn_wmma_f32_16x16x32_f16(false, ah, false, bh, (short)0, acc, false, false);
            acc = __builtin_amdgcn_wmma_f32_16x16x32_f16(false, ah, false, bl, (short)0, acc, false, false);
            acc = __builtin_amdgcn_wmma_f32_16x16x32_f16(false, al, false, bh, (short)0, acc, false, false);
            if (t) acc1 = acc; else acc0 = acc;
        }
    }

    // ---- Store: C/D layout -> VGPR v holds M = v + 8h, N = l15 ----
    const float bv = bias[0];
    const int xo = xblk + wcol + l15;
    if (xo < OUT_W) {
        #pragma unroll
        for (int v = 0; v < 8; ++v) {
            const int yo = y0 + v + 8 * h;
            if (yo < OUT_H) out[yo * OUT_W + xo] = acc0[v] + bv;
        }
        #pragma unroll
        for (int v = 0; v < 8; ++v) {
            const int yo = y0 + 16 + v + 8 * h;
            if (yo < OUT_H) out[yo * OUT_W + xo] = acc1[v] + bv;
        }
    }
}

extern "C" void kernel_launch(void* const* d_in, const int* in_sizes, int n_in,
                              void* d_out, int out_size, void* d_ws, size_t ws_size,
                              hipStream_t stream) {
    const float* X    = (const float*)d_in[0];   // 4096*4096 f32
    const float* Wk   = (const float*)d_in[1];   // 15*15 f32
    const float* bias = (const float*)d_in[2];   // 1 f32
    float* out        = (float*)d_out;           // 4082*4082 f32

    dim3 grid((OUT_W + BLK_COLS - 1) / BLK_COLS,   // 32
              (OUT_H + BLK_ROWS - 1) / BLK_ROWS);  // 128
    dim3 block(256);
    conv15x15_wmma_f16split<<<grid, block, 0, stream>>>(X, Wk, bias, out);
}
